// SlidingWindowAttention_25829933318182
// MI455X (gfx1250) — matrix operand order repaired and checked by reading, not compiled, not run on hardware
//
#include <hip/hip_runtime.h>
#include <hip/hip_bf16.h>

typedef __bf16 bf16_t;
typedef __attribute__((ext_vector_type(16))) __bf16 v16bf;
typedef __attribute__((ext_vector_type(8)))  __bf16 v8bf;
typedef __attribute__((ext_vector_type(4)))  __bf16 v4bf;
typedef __attribute__((ext_vector_type(8)))  float  v8f;

#define E_DIM   1024
#define H_NUM   16
#define HD      64
#define B_SZ    2
#define S_LEN   2048
#define MTOT    (B_SZ * S_LEN)   /* 4096 rows of x */
#define N_QKV   (3 * E_DIM)      /* 3072 */
#define HALF_WIN 128
#define NEG_BIG (-1.0e30f)

static __device__ __forceinline__ v16bf cat8(v8bf lo, v8bf hi) {
  return __builtin_shufflevector(lo, hi, 0,1,2,3,4,5,6,7,8,9,10,11,12,13,14,15);
}
static __device__ __forceinline__ v8f wmma_bf16(v16bf a, v16bf b, v8f c) {
  // D = A(16x32 bf16) x B(32x16 bf16) + C(16x16 f32)
  return __builtin_amdgcn_wmma_f32_16x16x32_bf16(false, a, false, b, (short)0, c,
                                                 false, false);
}

// ---------------------------------------------------------------- fp32 -> bf16
__global__ void cvt_f32_bf16(const float* __restrict__ src,
                             bf16_t* __restrict__ dst, int n4) {
  int i = blockIdx.x * blockDim.x + threadIdx.x;
  if (i < n4) {
    float4 f = ((const float4*)src)[i];
    v4bf o;
    o[0] = (bf16_t)f.x; o[1] = (bf16_t)f.y; o[2] = (bf16_t)f.z; o[3] = (bf16_t)f.w;
    *(v4bf*)(dst + 4 * (size_t)i) = o;
  }
}

// --------------------------------------------------- QKV projection (WMMA GEMM)
// C[m,n] = sum_k X[m,k] * W[n,k] + bias[n]; epilogue scatters into
// Q [b,h,s,d] (pre-scaled by 1/sqrt(hd)), K [b,h,s,d], V^T [b,h,d,s] (bf16).
__global__ __launch_bounds__(128) void qkv_gemm_wmma(
    const bf16_t* __restrict__ X, const bf16_t* __restrict__ W,
    const float* __restrict__ bias,
    bf16_t* __restrict__ Qp, bf16_t* __restrict__ Kp, bf16_t* __restrict__ Vt) {
  const int lane = threadIdx.x & 31;
  const int wave = threadIdx.x >> 5;
  const int half = lane >> 4;      // which 16-lane half
  const int lrow = lane & 15;
  const int mBase = blockIdx.x * 64 + wave * 16;
  const int nBase = blockIdx.y * 64;

  v8f acc[4] = {};
  const bf16_t* arow = X + (size_t)(mBase + lrow) * E_DIM;

  for (int k0 = 0; k0 < E_DIM; k0 += 32) {
    // A-frag: lane holds row (lane&15), 16 contiguous K at half*16
    v16bf a = cat8(*(const v8bf*)(arow + k0 + half * 16),
                   *(const v8bf*)(arow + k0 + half * 16 + 8));
#pragma unroll
    for (int t = 0; t < 4; ++t) {
      // B-frag: lane holds column (lane&15) = W row, 16 contiguous K at half*16
      const bf16_t* brow =
          W + (size_t)(nBase + t * 16 + lrow) * E_DIM + k0 + half * 16;
      acc[t] = wmma_bf16(a, *(const v16bf*)brow, acc[t]);
    }
  }
#pragma unroll
  for (int t = 0; t < 4; ++t) {
    const int n = nBase + t * 16 + lrow;
    const float bv = bias[n];
#pragma unroll
    for (int r = 0; r < 8; ++r) {
      const int m = mBase + r + half * 8;     // C-layout: VGPR r -> row r (+8)
      const int b = m >> 11, s = m & (S_LEN - 1);
      float v = acc[t][r] + bv;
      if (n < E_DIM) {                         // Q, fold in 1/sqrt(64)
        int h = n >> 6, d = n & 63;
        Qp[(((size_t)(b * H_NUM + h)) * S_LEN + s) * HD + d] = (bf16_t)(v * 0.125f);
      } else if (n < 2 * E_DIM) {              // K
        int nn = n - E_DIM, h = nn >> 6, d = nn & 63;
        Kp[(((size_t)(b * H_NUM + h)) * S_LEN + s) * HD + d] = (bf16_t)v;
      } else {                                 // V stored transposed [b,h,d,s]
        int nn = n - 2 * E_DIM, h = nn >> 6, d = nn & 63;
        Vt[(((size_t)(b * H_NUM + h)) * HD + d) * S_LEN + s] = (bf16_t)v;
      }
    }
  }
}

// -------------------------------------------- flash attention, inverted band mask
__global__ __launch_bounds__(128) void attn_wmma(
    const bf16_t* __restrict__ Qp, const bf16_t* __restrict__ Kp,
    const bf16_t* __restrict__ Vt, bf16_t* __restrict__ O) {
  __shared__ bf16_t pShare[4][16 * 32];       // per-wave P staging (C->A relayout)
  const int lane = threadIdx.x & 31;
  const int wave = threadIdx.x >> 5;
  const int half = lane >> 4;
  const int lrow = lane & 15;
  const int gw = blockIdx.x * 4 + wave;       // 0 .. B*H*(S/16)-1
  const int qt = gw & (S_LEN / 16 - 1);
  const int bh = gw >> 7;                     // b*16 + h
  const int i0 = qt * 16;

  const bf16_t* Qh = Qp + (size_t)bh * S_LEN * HD;
  const bf16_t* Kh = Kp + (size_t)bh * S_LEN * HD;
  const bf16_t* Vh = Vt + (size_t)bh * HD * S_LEN;

  v16bf qf[2];
  {
    const bf16_t* qrow = Qh + (size_t)(i0 + lrow) * HD;
#pragma unroll
    for (int c = 0; c < 2; ++c)
      qf[c] = cat8(*(const v8bf*)(qrow + c * 32 + half * 16),
                   *(const v8bf*)(qrow + c * 32 + half * 16 + 8));
  }

  v8f o0 = {}, o1 = {}, o2 = {}, o3 = {};     // O accumulator, hd 0..63
  float M8[8], L8[8];
#pragma unroll
  for (int r = 0; r < 8; ++r) { M8[r] = NEG_BIG; L8[r] = 0.0f; }
  const int ibase = i0 + half * 8;
  bf16_t* pw = pShare[wave];

  for (int j0 = 0; j0 < S_LEN; j0 += 32) {
    // chunk entirely inside masked band |i-j|<=128 -> skip (wave-uniform)
    if (j0 >= i0 - 113 && j0 <= i0 + 97) continue;
    const bool needMask = !(j0 <= i0 - 160 || j0 >= i0 + 144);

    // ---- S = Q K^T : 4 WMMAs over two 16-key tiles, hd split 2x32
    v8f sc[2] = {};
#pragma unroll
    for (int t = 0; t < 2; ++t) {
      const bf16_t* krow = Kh + (size_t)(j0 + t * 16 + lrow) * HD + half * 16;
      sc[t] = wmma_bf16(qf[0], *(const v16bf*)(krow), sc[t]);
      sc[t] = wmma_bf16(qf[1], *(const v16bf*)(krow + 32), sc[t]);
    }
    if (needMask) {
#pragma unroll
      for (int t = 0; t < 2; ++t) {
        const int j = j0 + t * 16 + lrow;
#pragma unroll
        for (int r = 0; r < 8; ++r) {
          const int d = (ibase + r) - j;
          if (d >= -HALF_WIN && d <= HALF_WIN) sc[t][r] = NEG_BIG;
        }
      }
    }
    // ---- online softmax: rows live across 16-lane halves
    float alpha[8];
#pragma unroll
    for (int r = 0; r < 8; ++r) {
      float mx = fmaxf(sc[0][r], sc[1][r]);
      mx = fmaxf(mx, __shfl_xor(mx, 1, 16));
      mx = fmaxf(mx, __shfl_xor(mx, 2, 16));
      mx = fmaxf(mx, __shfl_xor(mx, 4, 16));
      mx = fmaxf(mx, __shfl_xor(mx, 8, 16));
      const float mnew = fmaxf(M8[r], mx);
      alpha[r] = __expf(M8[r] - mnew);
      M8[r] = mnew;
    }
#pragma unroll
    for (int r = 0; r < 8; ++r) {
      sc[0][r] = __expf(sc[0][r] - M8[r]);
      sc[1][r] = __expf(sc[1][r] - M8[r]);
      float sum = sc[0][r] + sc[1][r];
      sum += __shfl_xor(sum, 1, 16);
      sum += __shfl_xor(sum, 2, 16);
      sum += __shfl_xor(sum, 4, 16);
      sum += __shfl_xor(sum, 8, 16);
      L8[r] = L8[r] * alpha[r] + sum;
    }
#pragma unroll
    for (int r = 0; r < 8; ++r) {
      o0[r] *= alpha[r]; o1[r] *= alpha[r]; o2[r] *= alpha[r]; o3[r] *= alpha[r];
    }
    // ---- relayout P (C-frag) -> A-frag via per-wave LDS
#pragma unroll
    for (int t = 0; t < 2; ++t)
#pragma unroll
      for (int r = 0; r < 8; ++r)
        pw[(r + half * 8) * 32 + t * 16 + lrow] = (bf16_t)sc[t][r];
    asm volatile("s_wait_dscnt 0x0" ::: "memory");   // in-order DS pipe, same wave
    v16bf pa = cat8(*(const v8bf*)(pw + lrow * 32 + half * 16),
                    *(const v8bf*)(pw + lrow * 32 + half * 16 + 8));
    // ---- O += P V : V^T columns are contiguous over keys
    o0 = wmma_bf16(pa, *(const v16bf*)(Vh + (size_t)(0 + lrow) * S_LEN + j0 + half * 16), o0);
    o1 = wmma_bf16(pa, *(const v16bf*)(Vh + (size_t)(16 + lrow) * S_LEN + j0 + half * 16), o1);
    o2 = wmma_bf16(pa, *(const v16bf*)(Vh + (size_t)(32 + lrow) * S_LEN + j0 + half * 16), o2);
    o3 = wmma_bf16(pa, *(const v16bf*)(Vh + (size_t)(48 + lrow) * S_LEN + j0 + half * 16), o3);
  }
  // ---- epilogue: normalize and scatter back to [b,s,h*64+d] bf16
  const int b = bh >> 4, h = bh & 15;
#pragma unroll
  for (int r = 0; r < 8; ++r) {
    const float inv = 1.0f / L8[r];
    const int s = i0 + r + half * 8;
    bf16_t* orow = O + ((size_t)(b * S_LEN + s)) * E_DIM + h * HD;
    orow[0  + lrow] = (bf16_t)(o0[r] * inv);
    orow[16 + lrow] = (bf16_t)(o1[r] * inv);
    orow[32 + lrow] = (bf16_t)(o2[r] * inv);
    orow[48 + lrow] = (bf16_t)(o3[r] * inv);
  }
}

// ------------------------------------------------ output projection (WMMA GEMM)
__global__ __launch_bounds__(128) void out_gemm_wmma(
    const bf16_t* __restrict__ A, const bf16_t* __restrict__ W,
    const float* __restrict__ bias, float* __restrict__ out) {
  const int lane = threadIdx.x & 31;
  const int wave = threadIdx.x >> 5;
  const int half = lane >> 4;
  const int lrow = lane & 15;
  const int mBase = blockIdx.x * 64 + wave * 16;
  const int nBase = blockIdx.y * 64;

  v8f acc[4] = {};
  const bf16_t* arow = A + (size_t)(mBase + lrow) * E_DIM;
  for (int k0 = 0; k0 < E_DIM; k0 += 32) {
    v16bf a = cat8(*(const v8bf*)(arow + k0 + half * 16),
                   *(const v8bf*)(arow + k0 + half * 16 + 8));
#pragma unroll
    for (int t = 0; t < 4; ++t) {
      const bf16_t* brow =
          W + (size_t)(nBase + t * 16 + lrow) * E_DIM + k0 + half * 16;
      acc[t] = wmma_bf16(a, *(const v16bf*)brow, acc[t]);
    }
  }
#pragma unroll
  for (int t = 0; t < 4; ++t) {
    const int n = nBase + t * 16 + lrow;
    const float bv = bias[n];
#pragma unroll
    for (int r = 0; r < 8; ++r) {
      const int m = mBase + r + half * 8;
      out[(size_t)m * E_DIM + n] = acc[t][r] + bv;
    }
  }
}

extern "C" void kernel_launch(void* const* d_in, const int* in_sizes, int n_in,
                              void* d_out, int out_size, void* d_ws, size_t ws_size,
                              hipStream_t stream) {
  const float* x    = (const float*)d_in[0];   // [B,S,E]
  const float* wqkv = (const float*)d_in[1];   // [3E,E]
  const float* bqkv = (const float*)d_in[2];   // [3E]
  const float* wout = (const float*)d_in[3];   // [E,E]
  const float* bout = (const float*)d_in[4];   // [E]

  bf16_t* p = (bf16_t*)d_ws;                   // ~48 MB of bf16 scratch
  bf16_t* xbf  = p; p += (size_t)MTOT * E_DIM;
  bf16_t* wqbf = p; p += (size_t)N_QKV * E_DIM;
  bf16_t* wobf = p; p += (size_t)E_DIM * E_DIM;
  bf16_t* Qp   = p; p += (size_t)MTOT * E_DIM;
  bf16_t* Kp   = p; p += (size_t)MTOT * E_DIM;
  bf16_t* Vt   = p; p += (size_t)MTOT * E_DIM;
  bf16_t* Obf  = p; p += (size_t)MTOT * E_DIM;

  const int nx = MTOT * E_DIM / 4, nw = N_QKV * E_DIM / 4, no = E_DIM * E_DIM / 4;
  cvt_f32_bf16<<<(nx + 255) / 256, 256, 0, stream>>>(x, xbf, nx);
  cvt_f32_bf16<<<(nw + 255) / 256, 256, 0, stream>>>(wqkv, wqbf, nw);
  cvt_f32_bf16<<<(no + 255) / 256, 256, 0, stream>>>(wout, wobf, no);

  qkv_gemm_wmma<<<dim3(MTOT / 64, N_QKV / 64), 128, 0, stream>>>(
      xbf, wqbf, bqkv, Qp, Kp, Vt);
  attn_wmma<<<(B_SZ * H_NUM * (S_LEN / 16)) / 4, 128, 0, stream>>>(Qp, Kp, Vt, Obf);
  out_gemm_wmma<<<dim3(MTOT / 64, E_DIM / 64), 128, 0, stream>>>(
      Obf, wobf, bout, (float*)d_out);
}
